// MultiHeadSelfAttention_69123203661967
// MI455X (gfx1250) — compile-verified, matrix operand order checked by reading, not run on hardware
//
#include <hip/hip_runtime.h>

// ---------------------------------------------------------------------------
// MultiHeadSelfAttention for MI455X (gfx1250, wave32, WMMA bf16)
// B=4, S=2048, D=1024, H=16, HD=64
// Pipeline: f32->bf16 convert | QKV GEMMs | flash attention | out-proj GEMM
// GEMMs: 128x128 block tile, K-step 64, double-buffered LDS, async G->LDS copy
// ---------------------------------------------------------------------------

#define B_  4
#define S_  2048
#define D_  1024
#define H_  16
#define HD_ 64
#define M_  (B_ * S_)   // 8192 rows for all projection GEMMs

typedef __bf16 bf16_t;
typedef __attribute__((ext_vector_type(16))) __bf16 bf16x16;
typedef __attribute__((ext_vector_type(8)))  __bf16 bf16x8;
typedef __attribute__((ext_vector_type(8)))  float  f32x8;
typedef __attribute__((ext_vector_type(4)))  int    i32x4;

// ---------------------------------------------------------------------------
// Async global->LDS copy (GLOBAL_LOAD_ASYNC_TO_LDS_B128, ASYNCcnt-tracked).
// Builtin takes (<4 x i32> addrspace(1)*, <4 x i32> addrspace(3)*, imm, imm).
// Guarded: falls back to a register copy if the builtins are unavailable.
// ---------------------------------------------------------------------------
#if defined(__has_builtin)
#if __has_builtin(__builtin_amdgcn_global_load_async_to_lds_b128) && \
    __has_builtin(__builtin_amdgcn_s_wait_asynccnt)
#define USE_ASYNC_LDS 1
#endif
#endif
#ifndef USE_ASYNC_LDS
#define USE_ASYNC_LDS 0
#endif

__device__ __forceinline__ void cp_async_b128(bf16_t* lds, const bf16_t* g) {
#if USE_ASYNC_LDS
    __builtin_amdgcn_global_load_async_to_lds_b128(
        (__attribute__((address_space(1))) i32x4*)(i32x4*)(void*)const_cast<bf16_t*>(g),
        (__attribute__((address_space(3))) i32x4*)(i32x4*)(void*)lds,
        /*offset=*/0, /*cpol=*/0);
#else
    *reinterpret_cast<bf16x8*>(lds) = *reinterpret_cast<const bf16x8*>(g);
#endif
}

__device__ __forceinline__ void async_wait_all() {
#if USE_ASYNC_LDS
    __builtin_amdgcn_s_wait_asynccnt(0);
#endif
}

// ---------------------------------------------------------------------------
// Fragment loader.
// ISA 7.12.2, 16-bit A-matrix 16x32 (and B == A-of-B^T):
//   lane 0-15 : M = lane,    K = kb+{0..7} and kb+16+{0..7} with kb = 0
//   lane 16-31: M = lane-16, same pattern with kb = 8
// => fragment = two contiguous 8-element (16B) chunks at rowPtr+kb, rowPtr+kb+16.
// ---------------------------------------------------------------------------
__device__ __forceinline__ bf16x16 load_frag_row(const bf16_t* rowPtr, int kb) {
    bf16x8 lo = *reinterpret_cast<const bf16x8*>(rowPtr + kb);
    bf16x8 hi = *reinterpret_cast<const bf16x8*>(rowPtr + kb + 16);
    return __builtin_shufflevector(lo, hi, 0, 1, 2, 3, 4, 5, 6, 7,
                                           8, 9, 10, 11, 12, 13, 14, 15);
}

__device__ __forceinline__ f32x8 wmma_bf16(bf16x16 a, bf16x16 b, f32x8 c) {
    return __builtin_amdgcn_wmma_f32_16x16x32_bf16(
        /*neg_a=*/false, a, /*neg_b=*/false, b,
        /*c_mod=*/(short)0, c, /*reuse_a=*/false, /*reuse_b=*/false);
}

// ---------------------------------------------------------------------------
// f32 -> bf16 conversion (grid-stride, float4 wide). n must be % 4 == 0.
// ---------------------------------------------------------------------------
__global__ void f32_to_bf16_kernel(const float* __restrict__ in,
                                   bf16_t* __restrict__ out, int n4) {
    int i = blockIdx.x * blockDim.x + threadIdx.x;
    int stride = gridDim.x * blockDim.x;
    const float4* in4 = reinterpret_cast<const float4*>(in);
    for (; i < n4; i += stride) {
        float4 v = in4[i];
        bf16_t* o = out + 4 * (size_t)i;
        o[0] = (bf16_t)v.x; o[1] = (bf16_t)v.y;
        o[2] = (bf16_t)v.z; o[3] = (bf16_t)v.w;
    }
}

// ---------------------------------------------------------------------------
// Tiled bf16 GEMM: C[M,N] = A[M,K] * W[K,N] (+ bias)
// Block tile 128x128, 256 threads = 8 waves, wave tile 64x32 (4x2 WMMA frags).
// K-step 64 (two WMMA k-substeps), double-buffered LDS, async A-tile staging.
// sA: row-major [m][k]. sW: transposed [n][k] so B-fragments are contiguous
// along K like A-fragments.
// ---------------------------------------------------------------------------
template <bool OUT_F32, bool ADD_BIAS>
__launch_bounds__(256)
__global__ void gemm_bf16_128(const bf16_t* __restrict__ A,
                              const bf16_t* __restrict__ W,
                              const float* __restrict__ bias,
                              float* __restrict__ Cf,
                              bf16_t* __restrict__ Cb,
                              int M, int N, int K) {
    __shared__ bf16_t sA[2][128][72];   // [m][k], 144B row stride (16B aligned)
    __shared__ bf16_t sW[2][128][72];   // [n][k] (transposed tile)

    const int t    = threadIdx.x;
    const int lane = t & 31;
    const int wave = t >> 5;
    const int bm   = blockIdx.x * 128;
    const int bn   = blockIdx.y * 128;
    const int wRow = (wave >> 2) * 64;   // 0 | 64
    const int wCol = (wave & 3) * 32;    // 0..96
    const int l15  = lane & 15;
    const int kb   = (lane & 16) ? 8 : 0;
    const int rOff = (lane & 16) ? 8 : 0;

    // Staging assignments (256 threads)
    const int aRow = t >> 1;           // 0..127
    const int aCol = (t & 1) * 16;     // 0 | 16
    const int wK   = t >> 3;           // 0..31 (+32 for second half)
    const int wN   = (t & 7) * 16;     // 0..112

    auto stageA = [&](int buf, int kk) {
        const bf16_t* gA = A + (size_t)(bm + aRow) * K + kk + aCol;
        cp_async_b128(&sA[buf][aRow][aCol],      gA);
        cp_async_b128(&sA[buf][aRow][aCol + 32], gA + 32);
    };
    auto stageW = [&](int buf, int kk) {
#pragma unroll
        for (int half = 0; half < 2; ++half) {
            const int kIdx = half * 32 + wK;
            const bf16_t* gW = W + (size_t)(kk + kIdx) * N + bn + wN;
            bf16x8 w0 = *reinterpret_cast<const bf16x8*>(gW);
            bf16x8 w1 = *reinterpret_cast<const bf16x8*>(gW + 8);
#pragma unroll
            for (int i = 0; i < 8; ++i) {
                sW[buf][wN + i][kIdx]     = w0[i];
                sW[buf][wN + 8 + i][kIdx] = w1[i];
            }
        }
    };

    const f32x8 fzero = {0.f, 0.f, 0.f, 0.f, 0.f, 0.f, 0.f, 0.f};
    f32x8 acc[4][2];
#pragma unroll
    for (int mi = 0; mi < 4; ++mi)
#pragma unroll
        for (int ni = 0; ni < 2; ++ni) acc[mi][ni] = fzero;

    // Prologue: stage first K-block into buffer 0.
    stageA(0, 0);
    stageW(0, 0);
    async_wait_all();
    __syncthreads();

    int buf = 0;
    for (int kk = 0; kk < K; kk += 64) {
        // Stage next K-block into the other buffer while computing on this one.
        if (kk + 64 < K) {
            stageA(buf ^ 1, kk + 64);
            stageW(buf ^ 1, kk + 64);
        }

#pragma unroll
        for (int ks = 0; ks < 2; ++ks) {
            const int kbs = kb + ks * 32;
            bf16x16 afr[4], bfr[2];
#pragma unroll
            for (int mi = 0; mi < 4; ++mi)
                afr[mi] = load_frag_row(&sA[buf][wRow + mi * 16 + l15][0], kbs);
#pragma unroll
            for (int ni = 0; ni < 2; ++ni)
                bfr[ni] = load_frag_row(&sW[buf][wCol + ni * 16 + l15][0], kbs);
#pragma unroll
            for (int mi = 0; mi < 4; ++mi)
#pragma unroll
                for (int ni = 0; ni < 2; ++ni)
                    acc[mi][ni] = wmma_bf16(afr[mi], bfr[ni], acc[mi][ni]);
        }

        async_wait_all();   // our async copies into buf^1 are complete
        __syncthreads();    // everyone done reading buf / writing buf^1
        buf ^= 1;
    }

    // --- epilogue: C layout VGPR r, lanes0-15 -> M=r, lanes16-31 -> M=r+8
#pragma unroll
    for (int mi = 0; mi < 4; ++mi) {
#pragma unroll
        for (int ni = 0; ni < 2; ++ni) {
            const int col = bn + wCol + ni * 16 + l15;
#pragma unroll
            for (int r = 0; r < 8; ++r) {
                const int row = bm + wRow + mi * 16 + r + rOff;
                float v = acc[mi][ni][r];
                if (ADD_BIAS) v += bias[col];
                if (OUT_F32)  Cf[(size_t)row * N + col] = v;
                else          Cb[(size_t)row * N + col] = (bf16_t)v;
            }
        }
    }
}

// ---------------------------------------------------------------------------
// Flash attention. Grid (S/128, H, B), 256 threads (8 waves, 16 queries each).
// Key loop in tiles of 32: scores (2 frags x 2 k-steps WMMA), online softmax,
// P repack via per-wave LDS, P x V (4 WMMA) into 16x64 f32 accumulator.
// K tile staged with async G->LDS copy; V tile transposed manually.
// ---------------------------------------------------------------------------
__launch_bounds__(256)
__global__ void flash_attn_kernel(const bf16_t* __restrict__ Q,
                                  const bf16_t* __restrict__ Kc,
                                  const bf16_t* __restrict__ Vc,
                                  const long long* __restrict__ mask,
                                  bf16_t* __restrict__ Oatt) {
    __shared__ bf16_t sK[32][72];      // [key][dim], 144B rows (16B aligned)
    __shared__ bf16_t sVT[64][40];     // [dim][key] transposed
    __shared__ bf16_t sP[8][16][40];   // per-wave probabilities [q][key]

    const int t    = threadIdx.x;
    const int lane = t & 31;
    const int wave = t >> 5;
    const int l15  = lane & 15;
    const int kb   = (lane & 16) ? 8 : 0;
    const int rOff = (lane & 16) ? 8 : 0;

    const int h  = blockIdx.y;
    const int b  = blockIdx.z;
    const int q0 = blockIdx.x * 128;

    // Per-wave Q fragments (16 queries x 64 dims = 2 k-step A-frags), from global.
    const int qRow = q0 + wave * 16 + l15;
    const bf16_t* qPtr = Q + ((size_t)(b * S_ + qRow)) * D_ + h * HD_;
    bf16x16 qf0 = load_frag_row(qPtr, kb);
    bf16x16 qf1 = load_frag_row(qPtr, kb + 32);

    const f32x8 fzero = {0.f, 0.f, 0.f, 0.f, 0.f, 0.f, 0.f, 0.f};
    f32x8 o[4];
#pragma unroll
    for (int c = 0; c < 4; ++c) o[c] = fzero;
    float mrun[8], lrun[8];
#pragma unroll
    for (int r = 0; r < 8; ++r) { mrun[r] = -1e30f; lrun[r] = 0.f; }

    const int sKey = t >> 3;          // 0..31
    const int sDim = (t & 7) * 8;     // 0..56

    for (int kt = 0; kt < S_; kt += 32) {
        // --- stage K tile (async, row-major) and V tile (manual transpose)
        const size_t kvBase = ((size_t)(b * S_ + kt + sKey)) * D_ + h * HD_ + sDim;
        cp_async_b128(&sK[sKey][sDim], Kc + kvBase);
        bf16x8 vv = *reinterpret_cast<const bf16x8*>(Vc + kvBase);
#pragma unroll
        for (int i = 0; i < 8; ++i) sVT[sDim + i][sKey] = vv[i];
        async_wait_all();
        __syncthreads();

        if (kt + 32 < S_) {
            __builtin_prefetch(Vc + kvBase + 32 * (size_t)D_, 0, 0);
        }

        // --- scores: S[16x32] = Q[16x64] * K^T, two 16-key fragments
        f32x8 sc[2];
#pragma unroll
        for (int nf = 0; nf < 2; ++nf) {
            const bf16_t* kRow = &sK[nf * 16 + l15][0];
            f32x8 s = fzero;
            s = wmma_bf16(qf0, load_frag_row(kRow, kb), s);
            s = wmma_bf16(qf1, load_frag_row(kRow, kb + 32), s);
            sc[nf] = s;
        }

        // --- mask (additive -inf per key column) + scale 1/sqrt(64)
        float madd[2];
#pragma unroll
        for (int nf = 0; nf < 2; ++nf) {
            long long mv = mask[(size_t)b * S_ + kt + nf * 16 + l15];
            madd[nf] = (mv != 0) ? 0.f : -__builtin_inff();
        }

        // --- online softmax; row M=r+rOff spans the 16 lanes of this half,
        //     so xor-butterfly over masks 1,2,4,8 reduces within the row.
        float corr[8];
#pragma unroll
        for (int r = 0; r < 8; ++r) {
            float s0 = sc[0][r] * 0.125f + madd[0];
            float s1 = sc[1][r] * 0.125f + madd[1];
            float mx = fmaxf(s0, s1);
#pragma unroll
            for (int off = 1; off <= 8; off <<= 1)
                mx = fmaxf(mx, __shfl_xor(mx, off, 32));
            float mnew = fmaxf(mrun[r], mx);
            float p0 = __expf(s0 - mnew);
            float p1 = __expf(s1 - mnew);
            corr[r] = __expf(mrun[r] - mnew);
            float rs = p0 + p1;
#pragma unroll
            for (int off = 1; off <= 8; off <<= 1)
                rs += __shfl_xor(rs, off, 32);
            lrun[r] = lrun[r] * corr[r] + rs;
            mrun[r] = mnew;
            sP[wave][r + rOff][l15]      = (bf16_t)p0;
            sP[wave][r + rOff][16 + l15] = (bf16_t)p1;
        }

        // --- rescale running output, then O += P[16x32] * V[32x64]
#pragma unroll
        for (int c = 0; c < 4; ++c)
#pragma unroll
            for (int r = 0; r < 8; ++r) o[c][r] *= corr[r];

        bf16x16 pf = load_frag_row(&sP[wave][l15][0], kb);
#pragma unroll
        for (int c = 0; c < 4; ++c) {
            bf16x16 vf = load_frag_row(&sVT[c * 16 + l15][0], kb);
            o[c] = wmma_bf16(pf, vf, o[c]);
        }
        __syncthreads();
    }

    // --- normalize and write attended [b, q, h*64+dim] as bf16
#pragma unroll
    for (int c = 0; c < 4; ++c) {
#pragma unroll
        for (int r = 0; r < 8; ++r) {
            const int q = q0 + wave * 16 + r + rOff;
            float val = o[c][r] / lrun[r];
            Oatt[((size_t)(b * S_ + q)) * D_ + h * HD_ + c * 16 + l15] = (bf16_t)val;
        }
    }
}

// ---------------------------------------------------------------------------
// Host launcher
// ---------------------------------------------------------------------------
extern "C" void kernel_launch(void* const* d_in, const int* in_sizes, int n_in,
                              void* d_out, int out_size, void* d_ws, size_t ws_size,
                              hipStream_t stream) {
    const float*     emb  = (const float*)d_in[0];
    const long long* mask = (const long long*)d_in[1];
    const float*     Wq   = (const float*)d_in[2];
    const float*     Wk   = (const float*)d_in[3];
    const float*     Wv   = (const float*)d_in[4];
    const float*     Wo   = (const float*)d_in[5];
    const float*     bo   = (const float*)d_in[6];
    float*           out  = (float*)d_out;

    const size_t MD = (size_t)M_ * D_;     // 8,388,608 elements
    const size_t DD = (size_t)D_ * D_;     // 1,048,576 elements

    char* ws = (char*)d_ws;
    size_t off = 0;
    auto carve = [&](size_t bytes) -> char* {
        char* p = ws + off;
        off += (bytes + 255) & ~(size_t)255;
        return p;
    };
    bf16_t* Xb  = (bf16_t*)carve(MD * 2);
    bf16_t* Wqb = (bf16_t*)carve(DD * 2);
    bf16_t* Wkb = (bf16_t*)carve(DD * 2);
    bf16_t* Wvb = (bf16_t*)carve(DD * 2);
    bf16_t* Wob = (bf16_t*)carve(DD * 2);
    bf16_t* Qb  = (bf16_t*)carve(MD * 2);
    bf16_t* Kb  = (bf16_t*)carve(MD * 2);
    bf16_t* Vb  = (bf16_t*)carve(MD * 2);
    bf16_t* Ab  = (bf16_t*)carve(MD * 2);
    (void)ws_size; (void)in_sizes; (void)n_in; (void)out_size;

    // 1) f32 -> bf16 conversions
    f32_to_bf16_kernel<<<2048, 256, 0, stream>>>(emb, Xb, (int)(MD / 4));
    f32_to_bf16_kernel<<<512, 256, 0, stream>>>(Wq, Wqb, (int)(DD / 4));
    f32_to_bf16_kernel<<<512, 256, 0, stream>>>(Wk, Wkb, (int)(DD / 4));
    f32_to_bf16_kernel<<<512, 256, 0, stream>>>(Wv, Wvb, (int)(DD / 4));
    f32_to_bf16_kernel<<<512, 256, 0, stream>>>(Wo, Wob, (int)(DD / 4));

    // 2) QKV projections: [8192,1024] x [1024,1024]
    dim3 gGemm(M_ / 128, D_ / 128);
    gemm_bf16_128<false, false><<<gGemm, 256, 0, stream>>>(
        Xb, Wqb, nullptr, nullptr, Qb, M_, D_, D_);
    gemm_bf16_128<false, false><<<gGemm, 256, 0, stream>>>(
        Xb, Wkb, nullptr, nullptr, Kb, M_, D_, D_);
    gemm_bf16_128<false, false><<<gGemm, 256, 0, stream>>>(
        Xb, Wvb, nullptr, nullptr, Vb, M_, D_, D_);

    // 3) Flash attention per (q-tile, head, batch)
    dim3 gAttn(S_ / 128, H_, B_);
    flash_attn_kernel<<<gAttn, 256, 0, stream>>>(Qb, Kb, Vb, mask, Ab);

    // 4) Output projection with bias, f32 output
    gemm_bf16_128<true, true><<<gGemm, 256, 0, stream>>>(
        Ab, Wob, bo, out, nullptr, M_, D_, D_);
}
